// UnetrUpBlockSLA_70360154243450
// MI455X (gfx1250) — matrix operand order repaired and checked by reading, not compiled, not run on hardware
//
#include <hip/hip_runtime.h>
#include <hip/hip_bf16.h>

typedef __attribute__((ext_vector_type(16))) _Float16 v16h;
typedef __attribute__((ext_vector_type(8)))  _Float16 v8h;
typedef __attribute__((ext_vector_type(8)))  float    v8f;

#define NB    16          // batch
#define CIN   256         // input channels to transp conv
#define CO    128         // Cout
#define XC    256         // concat channels (2*CO)
#define H1    48
#define P1    (48*48)     // 2304
#define H2    96
#define P2    (96*96)     // 9216
#define EPS_  1e-5f
#define SLOPE_ 0.01f

__device__ __forceinline__ v16h cat8(v8h lo, v8h hi) {
  return __builtin_shufflevector(lo, hi, 0,1,2,3,4,5,6,7,8,9,10,11,12,13,14,15);
}
__device__ __forceinline__ v8f wmma16(v16h a, v16h b, v8f c) {
  return __builtin_amdgcn_wmma_f32_16x16x32_f16(false, a, false, b, (short)0, c, false, false);
}
// CDNA5 async global->LDS copy (ASYNCcnt-tracked), 16B per lane.
__device__ __forceinline__ void async_b128_to_lds(const _Float16* gptr, const _Float16* lptr) {
  unsigned loff = (unsigned)(uintptr_t)lptr;  // low 32 bits of flat addr = LDS byte offset
  unsigned long long ga = (unsigned long long)(uintptr_t)gptr;
  asm volatile("global_load_async_to_lds_b128 %0, %1, off" :: "v"(loff), "v"(ga) : "memory");
}
__device__ __forceinline__ void wait_async0() {
  asm volatile("s_wait_asynccnt 0" ::: "memory");
}

// ---------- layout converts -------------------------------------------------
// inp NCHW f32 [16,256,48,48] -> NHWC f16 [16,48,48,256]
__global__ void cvt_inp_kernel(const float* __restrict__ inp, _Float16* __restrict__ dst) {
  size_t i = (size_t)blockIdx.x * blockDim.x + threadIdx.x;
  if (i >= (size_t)NB * P1 * CIN) return;
  int c = i % CIN; size_t p = i / CIN;
  int j = p % H1;  p /= H1;
  int ii = p % H1; int b = p / H1;
  dst[i] = (_Float16)inp[(((size_t)b * CIN + c) * H1 + ii) * H1 + j];
}

// skip NCHW f32 [16,128,96,96] -> x_h[..][128+c] (NHWC halves)
__global__ void cvt_skip_kernel(const float* __restrict__ skip, _Float16* __restrict__ x_h) {
  size_t i = (size_t)blockIdx.x * blockDim.x + threadIdx.x;
  if (i >= (size_t)NB * P2 * CO) return;
  int c = i % CO; size_t p = i / CO;
  int xx = p % H2; p /= H2;
  int y = p % H2;  int b = p / H2;
  x_h[(((size_t)b * P2) + y * H2 + xx) * XC + CO + c] =
      (_Float16)skip[(((size_t)b * CO + c) * H2 + y) * H2 + xx];
}

// wt [256,128,2,2] f32 -> wtT [4][128][256] f16   (dd = di*2+dj)
__global__ void cvt_wt_kernel(const float* __restrict__ wt, _Float16* __restrict__ wtT) {
  int i = blockIdx.x * blockDim.x + threadIdx.x;
  if (i >= 4 * CO * CIN) return;
  int c = i % CIN; int o = (i / CIN) % CO; int dd = i / (CIN * CO);
  int di = dd >> 1, dj = dd & 1;
  wtT[((size_t)dd * CO + o) * CIN + c] = (_Float16)wt[(((size_t)c * CO + o) * 2 + di) * 2 + dj];
}

// ---------- stage 0: transposed conv (4 GEMMs, one per (di,dj)) -------------
// LDS-staged weights (async double buffer) + pipelined A prefetch.
__global__ __launch_bounds__(256) void gemm_up_kernel(
    const _Float16* __restrict__ inp_h,  // [16][48][48][256]
    const _Float16* __restrict__ wtT,    // [4][128][256]
    _Float16* __restrict__ x_h)          // [16][96][96][256]
{
  constexpr int K_  = CIN;
  constexpr int KC  = 64;
  constexpr int NCH = K_ / KC;
  constexpr int RS  = KC + 8;
  __shared__ _Float16 wlds[2][CO * RS];

  int tid = threadIdx.x;
  int lane = tid & 31, wave = tid >> 5;
  int dd = blockIdx.y, di = dd >> 1, dj = dd & 1;
  int mt = blockIdx.x * 8 + wave;
  int m0 = mt * 16;
  int b = m0 / P1, pix = m0 % P1, i0 = pix / H1, j0 = pix % H1;
  int g = lane >> 4, ml = lane & 15;
  int ja = j0 + ml;
  v8f acc[8] = {};
  const _Float16* wb  = wtT + (size_t)dd * CO * CIN;
  const _Float16* ap0 = inp_h + (((size_t)b * H1 + i0) * H1 + ja) * CIN;

  auto stage = [&](int ch, int buf) {
    int kb = ch * KC;
#pragma unroll
    for (int it = 0; it < CO * (KC / 8) / 256; it++) {
      int t = tid + it * 256;
      int row = t / (KC / 8);
      int col = t % (KC / 8);
      async_b128_to_lds(wb + (size_t)row * K_ + kb + col * 8,
                        &wlds[buf][row * RS + col * 8]);
    }
  };
  auto loadA = [&](int ch, v16h* af) {
#pragma unroll
    for (int ks = 0; ks < KC / 32; ks++) {
      const _Float16* ap = ap0 + ch * KC + ks * 32 + 8 * g;
      af[ks] = cat8(*(const v8h*)ap, *(const v8h*)(ap + 16));
    }
  };

  v16h acur[KC / 32], anext[KC / 32];
  stage(0, 0);
  loadA(0, acur);
  wait_async0();
  __syncthreads();

  for (int ch = 0; ch < NCH; ch++) {
    int buf = ch & 1;
    if (ch + 1 < NCH) { stage(ch + 1, buf ^ 1); loadA(ch + 1, anext); }
#pragma unroll
    for (int ks = 0; ks < KC / 32; ks++)
#pragma unroll
      for (int nt = 0; nt < 8; nt++) {
        const _Float16* lp = &wlds[buf][(nt * 16 + ml) * RS + ks * 32 + 8 * g];
        acc[nt] = wmma16(acur[ks], cat8(*(const v8h*)lp, *(const v8h*)(lp + 16)), acc[nt]);
      }
#pragma unroll
    for (int ks = 0; ks < KC / 32; ks++) acur[ks] = anext[ks];
    if (ch + 1 < NCH) { wait_async0(); __syncthreads(); }
  }

#pragma unroll
  for (int nt = 0; nt < 8; nt++)
#pragma unroll
    for (int r = 0; r < 8; r++) {
      int j = j0 + r + 8 * g;
      x_h[(((size_t)b * H2 + 2 * i0 + di) * H2 + 2 * j + dj) * XC + nt * 16 + ml] =
          (_Float16)acc[nt][r];
    }
}

// ---------- pooling / gating / weight mixing --------------------------------
__global__ void pooled_kernel(const _Float16* __restrict__ src, float* __restrict__ pooled, int C) {
  int b = blockIdx.x, c = threadIdx.x;
  float s = 0.f;
  for (int p = 0; p < P2; p++) s += (float)src[((size_t)b * P2 + p) * C + c];
  pooled[b * C + c] = s / (float)P2;
}

__global__ void gate_kernel(const float* __restrict__ pooled, const float* __restrict__ Wg,
                            const float* __restrict__ bg, float* __restrict__ gout, int C) {
  int b = threadIdx.x;
  if (b >= NB) return;
  float z[4];
  for (int k = 0; k < 4; k++) {
    float s = bg[k];
    for (int c = 0; c < C; c++) s += pooled[b * C + c] * Wg[k * C + c];
    z[k] = s;
  }
  float zs[4] = {z[0], z[1], z[2], z[3]};
  for (int a = 0; a < 3; a++)
    for (int i = 0; i < 3 - a; i++)
      if (zs[i] < zs[i + 1]) { float t = zs[i]; zs[i] = zs[i + 1]; zs[i + 1] = t; }
  float cs = 0.f, csk = 0.f; int kmax = 0;
  for (int i = 0; i < 4; i++) {
    cs += zs[i];
    if (1.0f + (float)(i + 1) * zs[i] > cs) { kmax = i + 1; csk = cs; }
  }
  float tau = (csk - 1.0f) / (float)kmax;
  for (int k = 0; k < 4; k++) gout[b * 4 + k] = fmaxf(z[k] - tau, 0.f);
}

// Weff[b][o][t*Cin+c] = conv_w + (t==center ? sum_k g[b][k]*A[k][o][c] : 0), f16
__global__ void build_weff_kernel(const float* __restrict__ convw, const float* __restrict__ Amix,
                                  const float* __restrict__ g, _Float16* __restrict__ Weff,
                                  int Cin_, int ntap, size_t n) {
  size_t i = (size_t)blockIdx.x * blockDim.x + threadIdx.x;
  if (i >= n) return;
  int c = i % Cin_; size_t t2 = i / Cin_;
  int t = t2 % ntap; size_t t3 = t2 / ntap;
  int o = t3 % CO; int b = t3 / CO;
  float base = (ntap == 9) ? convw[((size_t)o * Cin_ + c) * 9 + t]
                           : convw[(size_t)o * Cin_ + c];
  float add = 0.f;
  if (t == ntap / 2)
    for (int k = 0; k < 4; k++) add += g[b * 4 + k] * Amix[((size_t)k * CO + o) * Cin_ + c];
  Weff[(((size_t)b * CO + o) * ntap + t) * Cin_ + c] = (_Float16)(base + add);
}

// ---------- main implicit-GEMM (3x3 or 1x1), per-sample weights -------------
// Weights staged to LDS with double-buffered async copies; A fragments for the
// next K-chunk prefetched during the current WMMA burst.
template <int C_, bool IS3X3>
__global__ __launch_bounds__(256) void gemm_conv_kernel(
    const _Float16* __restrict__ src,   // [16][96][96][C_]
    const _Float16* __restrict__ Weff,  // [16][128][K_]
    float* __restrict__ out)            // [16][9216][128]
{
  constexpr int K_  = IS3X3 ? 9 * C_ : C_;
  constexpr int KC  = 64;               // K-chunk staged in LDS
  constexpr int NCH = K_ / KC;
  constexpr int RS  = KC + 8;           // padded row stride (halves)
  __shared__ _Float16 wlds[2][CO * RS];

  int tid = threadIdx.x;
  int lane = tid & 31, wave = tid >> 5;
  int mt = blockIdx.x * 8 + wave;
  int m0 = mt * 16;
  int b = m0 / P2, pix = m0 % P2, y = pix / H2, x0 = pix % H2;
  int g = lane >> 4, ml = lane & 15;
  int xa = x0 + ml;
  v8f acc[8] = {};
  const _Float16* wbase = Weff + (size_t)b * CO * K_;

  auto stage = [&](int ch, int buf) {
    int kb = ch * KC;
#pragma unroll
    for (int it = 0; it < CO * (KC / 8) / 256; it++) {
      int t = tid + it * 256;
      int row = t / (KC / 8);
      int col = t % (KC / 8);  // units of 8 halves (16B)
      async_b128_to_lds(wbase + (size_t)row * K_ + kb + col * 8,
                        &wlds[buf][row * RS + col * 8]);
    }
  };
  auto loadA = [&](int ch, v16h* af) {
#pragma unroll
    for (int ks = 0; ks < KC / 32; ks++) {
      int kb = ch * KC + ks * 32;
      int yy, xx, coff;
      if (IS3X3) {
        int tap = kb / C_;
        coff = kb % C_;
        yy = y + tap / 3 - 1;
        xx = xa + tap % 3 - 1;
      } else { coff = kb; yy = y; xx = xa; }
      if (yy >= 0 && yy < H2 && xx >= 0 && xx < H2) {
        const _Float16* ap = src + (((size_t)b * P2 + yy * H2 + xx) * C_) + coff + 8 * g;
        af[ks] = cat8(*(const v8h*)ap, *(const v8h*)(ap + 16));
      } else {
        v8h z = {};
        af[ks] = cat8(z, z);
      }
    }
  };

  v16h acur[KC / 32], anext[KC / 32];
  stage(0, 0);
  loadA(0, acur);
  wait_async0();
  __syncthreads();

  for (int ch = 0; ch < NCH; ch++) {
    int buf = ch & 1;
    if (ch + 1 < NCH) { stage(ch + 1, buf ^ 1); loadA(ch + 1, anext); }

#pragma unroll
    for (int ks = 0; ks < KC / 32; ks++)
#pragma unroll
      for (int nt = 0; nt < 8; nt++) {
        const _Float16* lp = &wlds[buf][(nt * 16 + ml) * RS + ks * 32 + 8 * g];
        v16h bv = cat8(*(const v8h*)lp, *(const v8h*)(lp + 16));
        acc[nt] = wmma16(acur[ks], bv, acc[nt]);
      }
#pragma unroll
    for (int ks = 0; ks < KC / 32; ks++) acur[ks] = anext[ks];

    if (ch + 1 < NCH) {
      wait_async0();
      __syncthreads();
    }
  }

#pragma unroll
  for (int nt = 0; nt < 8; nt++)
#pragma unroll
    for (int r = 0; r < 8; r++)
      out[((size_t)b * P2 + y * H2 + x0 + r + 8 * g) * CO + nt * 16 + ml] = acc[nt][r];
}

// ---------- instnorm stats / apply ------------------------------------------
__global__ void stats_kernel(const float* __restrict__ src, float* __restrict__ mean,
                             float* __restrict__ rstd) {
  int b = blockIdx.x, c = threadIdx.x;  // blockDim = 128
  float s = 0.f, s2 = 0.f;
  for (int p = 0; p < P2; p++) {
    float v = src[((size_t)b * P2 + p) * CO + c];
    s += v; s2 += v * v;
  }
  float m = s / (float)P2;
  float var = s2 / (float)P2 - m * m;
  mean[b * CO + c] = m;
  rstd[b * CO + c] = rsqrtf(var + EPS_);
}

__global__ void norm_lrelu_f16_kernel(const float* __restrict__ src, const float* __restrict__ mean,
                                      const float* __restrict__ rstd, _Float16* __restrict__ dst) {
  size_t i = (size_t)blockIdx.x * blockDim.x + threadIdx.x;
  if (i >= (size_t)NB * P2 * CO) return;
  int c = i % CO;
  int b = i / ((size_t)P2 * CO);
  float v = (src[i] - mean[b * CO + c]) * rstd[b * CO + c];
  v = v >= 0.f ? v : SLOPE_ * v;
  dst[i] = (_Float16)v;
}

__global__ void final_kernel(const float* __restrict__ out2, const float* __restrict__ res,
                             const float* __restrict__ m2, const float* __restrict__ r2,
                             const float* __restrict__ m3, const float* __restrict__ r3,
                             float* __restrict__ out) {
  size_t i = (size_t)blockIdx.x * blockDim.x + threadIdx.x;  // NCHW index
  if (i >= (size_t)NB * CO * P2) return;
  int p = i % P2; int o = (i / P2) % CO; int b = i / ((size_t)P2 * CO);
  size_t si = ((size_t)b * P2 + p) * CO + o;
  float v = (out2[si] - m2[b * CO + o]) * r2[b * CO + o] +
            (res[si]  - m3[b * CO + o]) * r3[b * CO + o];
  out[i] = v >= 0.f ? v : SLOPE_ * v;
}

// ---------- launch ----------------------------------------------------------
extern "C" void kernel_launch(void* const* d_in, const int* in_sizes, int n_in,
                              void* d_out, int out_size, void* d_ws, size_t ws_size,
                              hipStream_t stream) {
  const float* inp     = (const float*)d_in[0];
  const float* skip    = (const float*)d_in[1];
  const float* wt      = (const float*)d_in[2];
  const float* conv1_w = (const float*)d_in[3];
  const float* Wg1     = (const float*)d_in[4];
  const float* bg1     = (const float*)d_in[5];
  const float* A1      = (const float*)d_in[6];
  const float* conv2_w = (const float*)d_in[7];
  const float* Wg2     = (const float*)d_in[8];
  const float* bg2     = (const float*)d_in[9];
  const float* A2      = (const float*)d_in[10];
  const float* conv3_w = (const float*)d_in[11];
  const float* Wg3     = (const float*)d_in[12];
  const float* bg3     = (const float*)d_in[13];
  const float* A3      = (const float*)d_in[14];
  float* out = (float*)d_out;

  char* ws = (char*)d_ws;
  size_t off = 0;
  auto carve = [&](size_t bytes) -> char* {
    off = (off + 255) & ~(size_t)255;
    char* p = ws + off;
    off += bytes;
    return p;
  };
  _Float16* inp_h  = (_Float16*)carve((size_t)NB * P1 * CIN * 2);
  _Float16* x_h    = (_Float16*)carve((size_t)NB * P2 * XC * 2);
  _Float16* wtT    = (_Float16*)carve((size_t)4 * CO * CIN * 2);
  float*    pool1  = (float*)carve((size_t)NB * XC * 4);
  float*    pool2  = (float*)carve((size_t)NB * CO * 4);
  float*    g1     = (float*)carve(NB * 4 * 4);
  float*    g2     = (float*)carve(NB * 4 * 4);
  float*    g3     = (float*)carve(NB * 4 * 4);
  _Float16* W1eff  = (_Float16*)carve((size_t)NB * CO * 9 * XC * 2);
  _Float16* W2eff  = (_Float16*)carve((size_t)NB * CO * 9 * CO * 2);
  _Float16* W3eff  = (_Float16*)carve((size_t)NB * CO * XC * 2);
  float*    out1   = (float*)carve((size_t)NB * P2 * CO * 4);   // reused as `res`
  _Float16* x2_h   = (_Float16*)carve((size_t)NB * P2 * CO * 2);
  float*    out2   = (float*)carve((size_t)NB * P2 * CO * 4);
  float*    mean1  = (float*)carve(NB * CO * 4);
  float*    rstd1  = (float*)carve(NB * CO * 4);
  float*    mean2  = (float*)carve(NB * CO * 4);
  float*    rstd2  = (float*)carve(NB * CO * 4);
  float*    mean3  = (float*)carve(NB * CO * 4);
  float*    rstd3  = (float*)carve(NB * CO * 4);
  float*    res    = out1;  // out1 is dead after x2_h is written

  // stage 0: convert layouts, transposed conv into x_h
  cvt_inp_kernel<<<(NB * P1 * CIN + 255) / 256, 256, 0, stream>>>(inp, inp_h);
  cvt_skip_kernel<<<(NB * P2 * CO + 255) / 256, 256, 0, stream>>>(skip, x_h);
  cvt_wt_kernel<<<(4 * CO * CIN + 255) / 256, 256, 0, stream>>>(wt, wtT);
  gemm_up_kernel<<<dim3(P1 * NB / 16 / 8, 4), 256, 0, stream>>>(inp_h, wtT, x_h);

  // stage 1: gates from pooled x; mixed effective weights for conv1 and 1x1 res
  pooled_kernel<<<NB, XC, 0, stream>>>(x_h, pool1, XC);
  gate_kernel<<<1, 32, 0, stream>>>(pool1, Wg1, bg1, g1, XC);
  gate_kernel<<<1, 32, 0, stream>>>(pool1, Wg3, bg3, g3, XC);
  build_weff_kernel<<<((size_t)NB * CO * 9 * XC + 255) / 256, 256, 0, stream>>>(
      conv1_w, A1, g1, W1eff, XC, 9, (size_t)NB * CO * 9 * XC);
  build_weff_kernel<<<((size_t)NB * CO * XC + 255) / 256, 256, 0, stream>>>(
      conv3_w, A3, g3, W3eff, XC, 1, (size_t)NB * CO * XC);

  // stage 2: conv1 (implicit GEMM, K=2304) -> out1 ; instnorm+lrelu -> x2_h
  gemm_conv_kernel<XC, true><<<NB * P2 / 16 / 8, 256, 0, stream>>>(x_h, W1eff, out1);
  stats_kernel<<<NB, CO, 0, stream>>>(out1, mean1, rstd1);
  norm_lrelu_f16_kernel<<<((size_t)NB * P2 * CO + 255) / 256, 256, 0, stream>>>(
      out1, mean1, rstd1, x2_h);

  // stage 3: gate2 from pooled x2; conv2 (K=1152) -> out2 ; instnorm stats
  pooled_kernel<<<NB, CO, 0, stream>>>(x2_h, pool2, CO);
  gate_kernel<<<1, 32, 0, stream>>>(pool2, Wg2, bg2, g2, CO);
  build_weff_kernel<<<((size_t)NB * CO * 9 * CO + 255) / 256, 256, 0, stream>>>(
      conv2_w, A2, g2, W2eff, CO, 9, (size_t)NB * CO * 9 * CO);
  gemm_conv_kernel<CO, true><<<NB * P2 / 16 / 8, 256, 0, stream>>>(x2_h, W2eff, out2);
  stats_kernel<<<NB, CO, 0, stream>>>(out2, mean2, rstd2);

  // stage 4: residual 1x1 (K=256) -> res (aliases out1) ; instnorm stats
  gemm_conv_kernel<XC, false><<<NB * P2 / 16 / 8, 256, 0, stream>>>(x_h, W3eff, res);
  stats_kernel<<<NB, CO, 0, stream>>>(res, mean3, rstd3);

  // stage 5: lrelu(norm(out2) + norm(res)) -> NCHW output
  final_kernel<<<((size_t)NB * CO * P2 + 255) / 256, 256, 0, stream>>>(
      out2, res, mean2, rstd2, mean3, rstd3, out);
}
